// GeoMoE_84920093376653
// MI455X (gfx1250) — matrix operand order repaired and checked by reading, not compiled, not run on hardware
//
#include <hip/hip_runtime.h>
#include <hip/hip_bf16.h>
#include <math.h>

// ---------------- problem constants ----------------
#define T_TOK 4096
#define DIM   1024
#define HID   768
#define NEXP  23
#define KTOP  3
#define NMAT  (NEXP + 1)               // experts + shared "expert 23"
#define SLOTS (T_TOK * KTOP + T_TOK)   // 16384
#define MAXTILE 64                     // worst case: one expert gets all tokens
#define JOBS  (NMAT * MAXTILE)

// ---------------- workspace layout (bytes, 256-aligned) ----------------
#define OFF_XBF   0ull
#define OFF_W1T   8388608ull
#define OFF_W2T   46137344ull
#define OFF_H     83886080ull
#define OFF_TIDX  109051904ull
#define OFF_TW    109101056ull
#define OFF_TOKID 109150208ull
#define OFF_TOKW  109215744ull
#define OFF_META  109281280ull

typedef __attribute__((ext_vector_type(16))) __bf16 bf16x16;
typedef __attribute__((ext_vector_type(8)))  float  floatx8;

union BV { bf16x16 v; uint4 q[2]; };

__device__ __forceinline__ float gelu_exact(float x) {
    return 0.5f * x * (1.0f + erff(x * 0.70710678118654752440f));
}

// ---- 64(M) x 64(N) x KTOT WMMA tile, one wave32 -----------------------------
// A (16-bit 16x32): lane r=L&15 is row M; halves at k = kk + 8*hi + {0..7} and +16.
// B (16-bit 32x16): lane r=L&15 is col N; halves at k = kk + 8*hi + {0..7} and +16.
// Per K-step: 8 A b128 loads + 8 B b128 loads feed 16 WMMAs (1.0 loads/WMMA).
template<int LDB, int KTOT>
__device__ __forceinline__ void wmma_tile64(const __bf16* const (&arow)[4],
                                            const __bf16* __restrict__ bcol,
                                            floatx8 (&acc)[4][4]) {
    for (int kk = 0; kk < KTOT; kk += 32) {
        BV a[4], b[4];
        #pragma unroll
        for (int m = 0; m < 4; ++m) {
            a[m].q[0] = *(const uint4*)(arow[m] + kk);
            a[m].q[1] = *(const uint4*)(arow[m] + kk + 16);
        }
        #pragma unroll
        for (int n = 0; n < 4; ++n) {
            b[n].q[0] = *(const uint4*)(bcol + (size_t)n * 16 * LDB + kk);
            b[n].q[1] = *(const uint4*)(bcol + (size_t)n * 16 * LDB + kk + 16);
        }
        #pragma unroll
        for (int m = 0; m < 4; ++m)
            #pragma unroll
            for (int n = 0; n < 4; ++n)
                acc[m][n] = __builtin_amdgcn_wmma_f32_16x16x32_bf16(
                    false, a[m].v, false, b[n].v, (short)0, acc[m][n], false, false);
    }
}

// ---------------- x -> bf16 ----------------
__global__ __launch_bounds__(256) void GeoMoE_cvt(const float* __restrict__ src,
                                                  __bf16* __restrict__ dst, int n) {
    int i = (blockIdx.x * 256 + threadIdx.x) * 4;
    if (i + 3 < n) {
        float4 f = *(const float4*)(src + i);
        dst[i + 0] = (__bf16)f.x;
        dst[i + 1] = (__bf16)f.y;
        dst[i + 2] = (__bf16)f.z;
        dst[i + 3] = (__bf16)f.w;
    }
}

// ---------------- [mat][R][C] f32 -> [mat][C][R] bf16 (tiled transpose) ----------------
__global__ __launch_bounds__(256) void GeoMoE_transpose(const float* __restrict__ src,
                                                        __bf16* __restrict__ dst,
                                                        int R, int C) {
    __shared__ float tile[32][33];
    size_t mat = blockIdx.z;
    src += mat * (size_t)R * C;
    dst += mat * (size_t)R * C;
    int c0 = blockIdx.x * 32, r0 = blockIdx.y * 32;
    #pragma unroll
    for (int j = 0; j < 4; ++j) {
        int r = r0 + threadIdx.y + j * 8;
        tile[threadIdx.y + j * 8][threadIdx.x] = src[(size_t)r * C + c0 + threadIdx.x];
    }
    __syncthreads();
    #pragma unroll
    for (int j = 0; j < 4; ++j) {
        int c = c0 + threadIdx.y + j * 8;
        dst[(size_t)c * R + r0 + threadIdx.x] = (__bf16)tile[threadIdx.x][threadIdx.y + j * 8];
    }
}

// ---------------- gating: one wave32 per token ----------------
__global__ __launch_bounds__(32) void GeoMoE_gate(const float* __restrict__ x,
                                                  const float* __restrict__ gW,
                                                  const float* __restrict__ gb,
                                                  const float* __restrict__ ebias,
                                                  int* __restrict__ top_idx,
                                                  float* __restrict__ top_w,
                                                  int* __restrict__ counts,
                                                  float* __restrict__ P_sum,
                                                  int* __restrict__ tok_id,
                                                  float* __restrict__ tok_w) {
    int t = blockIdx.x;
    int lane = threadIdx.x;
    float s = 0.0f;
    if (lane < NEXP) {
        const float* xr = x + (size_t)t * DIM;
        #pragma unroll 4
        for (int d = 0; d < DIM; ++d) s += xr[d] * gW[d * NEXP + lane];
        s += gb[lane];
    }
    float g = (lane < NEXP) ? 1.0f / (1.0f + expf(-s)) : 0.0f;

    float gs = g;
    #pragma unroll
    for (int m = 16; m >= 1; m >>= 1) gs += __shfl_xor(gs, m);
    if (lane < NEXP) atomicAdd(&P_sum[lane], g / gs);

    float sel = (lane < NEXP) ? g + ebias[lane] : -INFINITY;
    int   idx[KTOP];
    float wv[KTOP];
    #pragma unroll
    for (int k = 0; k < KTOP; ++k) {
        float v = sel; int ii = lane;
        #pragma unroll
        for (int m = 16; m >= 1; m >>= 1) {
            float ov = __shfl_xor(v, m);
            int   oi = __shfl_xor(ii, m);
            if (ov > v || (ov == v && oi < ii)) { v = ov; ii = oi; }
        }
        idx[k] = ii;
        wv[k]  = __shfl(g, ii);
        if (lane == ii) sel = -INFINITY;
    }
    float wsum = wv[0] + wv[1] + wv[2];
    if (lane == 0) {
        #pragma unroll
        for (int k = 0; k < KTOP; ++k) {
            top_idx[t * KTOP + k] = idx[k];
            top_w[t * KTOP + k]   = wv[k] / wsum;
            atomicAdd(&counts[idx[k]], 1);
        }
        tok_id[T_TOK * KTOP + t] = t;
        tok_w[T_TOK * KTOP + t]  = 1.0f;
    }
}

// ---------------- scan counts -> segments, aux loss, counts out ----------------
__global__ void GeoMoE_scan(const int* __restrict__ counts,
                            const float* __restrict__ P_sum,
                            int* __restrict__ seg_start, int* __restrict__ seg_len,
                            int* __restrict__ cursor,
                            float* __restrict__ out) {
    if (threadIdx.x != 0) return;
    int acc = 0;
    for (int e = 0; e < NEXP; ++e) {
        seg_start[e] = acc;
        seg_len[e]   = counts[e];
        cursor[e]    = 0;
        acc += counts[e];
    }
    seg_start[NEXP] = T_TOK * KTOP;
    seg_len[NEXP]   = T_TOK;
    float aux = 0.0f;
    for (int e = 0; e < NEXP; ++e) {
        float P = P_sum[e] / (float)T_TOK;
        float F = (float)NEXP * (float)counts[e] / ((float)KTOP * (float)T_TOK);
        aux += P * F;
    }
    out[(size_t)T_TOK * DIM] = aux;
    int* cout = (int*)(out + (size_t)T_TOK * DIM + 1);
    for (int e = 0; e < NEXP; ++e) cout[e] = counts[e];
}

// ---------------- scatter tokens into expert segments ----------------
__global__ __launch_bounds__(256) void GeoMoE_scatter(const int* __restrict__ top_idx,
                                                      const float* __restrict__ top_w,
                                                      const int* __restrict__ seg_start,
                                                      int* __restrict__ cursor,
                                                      int* __restrict__ tok_id,
                                                      float* __restrict__ tok_w) {
    int i = blockIdx.x * 256 + threadIdx.x;
    if (i >= T_TOK * KTOP) return;
    int t = i / KTOP;
    int e = top_idx[i];
    int pos = atomicAdd(&cursor[e], 1);
    int slot = seg_start[e] + pos;
    tok_id[slot] = t;
    tok_w[slot]  = top_w[i];
}

// ---------------- FFN layer 1: h = gelu(x_e @ W1[e] + b1[e]) ----------------
// one wave per block, 64(M) x 64(N) tile
__global__ __launch_bounds__(32) void GeoMoE_ffn1(const __bf16* __restrict__ xbf,
                                                  const __bf16* __restrict__ w1t,
                                                  const float* __restrict__ b1,
                                                  const float* __restrict__ sb1,
                                                  const int* __restrict__ seg_start,
                                                  const int* __restrict__ seg_len,
                                                  const int* __restrict__ tok_id,
                                                  __bf16* __restrict__ h) {
    int e    = blockIdx.x >> 6;
    int tile = blockIdx.x & 63;
    int len  = seg_len[e];
    if (tile * 64 >= len) return;
    int base = seg_start[e];
    int nb   = blockIdx.y * 64;
    int lane = threadIdx.x & 31;
    int r = lane & 15, hi = lane >> 4;

    const __bf16* arow[4];
    #pragma unroll
    for (int m = 0; m < 4; ++m) {
        int mrow = tile * 64 + m * 16 + r;
        int slot = base + (mrow < len ? mrow : 0);
        int grow = tok_id[slot];
        arow[m] = xbf + (size_t)grow * DIM + hi * 8;
    }
    const __bf16* bcol = w1t + (size_t)e * HID * DIM + (size_t)(nb + r) * DIM + hi * 8;

    floatx8 acc[4][4];
    #pragma unroll
    for (int m = 0; m < 4; ++m)
        #pragma unroll
        for (int n = 0; n < 4; ++n) acc[m][n] = (floatx8)0.0f;

    wmma_tile64<DIM, DIM>(arow, bcol, acc);

    const float* bptr = (e < NEXP) ? (b1 + (size_t)e * HID) : sb1;
    float bv[4];
    #pragma unroll
    for (int n = 0; n < 4; ++n) bv[n] = bptr[nb + n * 16 + r];

    #pragma unroll
    for (int m = 0; m < 4; ++m) {
        #pragma unroll
        for (int n = 0; n < 4; ++n) {
            int col = nb + n * 16 + r;
            #pragma unroll
            for (int v = 0; v < 8; ++v) {
                int ml = tile * 64 + m * 16 + v + 8 * hi;
                if (ml < len) {
                    float val = gelu_exact(acc[m][n][v] + bv[n]);
                    h[(size_t)(base + ml) * HID + col] = (__bf16)val;
                }
            }
        }
    }
}

// ---------------- FFN layer 2: out[tok] += w * (h_e @ W2[e] + b2[e]) ----------------
__global__ __launch_bounds__(32) void GeoMoE_ffn2(const __bf16* __restrict__ h,
                                                  const __bf16* __restrict__ w2t,
                                                  const float* __restrict__ b2,
                                                  const float* __restrict__ sb2,
                                                  const int* __restrict__ seg_start,
                                                  const int* __restrict__ seg_len,
                                                  const int* __restrict__ tok_id,
                                                  const float* __restrict__ tok_w,
                                                  float* __restrict__ out) {
    int e    = blockIdx.x >> 6;
    int tile = blockIdx.x & 63;
    int len  = seg_len[e];
    if (tile * 64 >= len) return;
    int base = seg_start[e];
    int nb   = blockIdx.y * 64;
    int lane = threadIdx.x & 31;
    int r = lane & 15, hi = lane >> 4;

    const __bf16* arow[4];
    #pragma unroll
    for (int m = 0; m < 4; ++m) {
        int mrow = tile * 64 + m * 16 + r;
        int slot = base + (mrow < len ? mrow : 0);
        arow[m] = h + (size_t)slot * HID + hi * 8;
    }
    const __bf16* bcol = w2t + (size_t)e * DIM * HID + (size_t)(nb + r) * HID + hi * 8;

    floatx8 acc[4][4];
    #pragma unroll
    for (int m = 0; m < 4; ++m)
        #pragma unroll
        for (int n = 0; n < 4; ++n) acc[m][n] = (floatx8)0.0f;

    wmma_tile64<HID, HID>(arow, bcol, acc);

    const float* bptr = (e < NEXP) ? (b2 + (size_t)e * DIM) : sb2;
    float bv[4];
    #pragma unroll
    for (int n = 0; n < 4; ++n) bv[n] = bptr[nb + n * 16 + r];

    #pragma unroll
    for (int m = 0; m < 4; ++m) {
        int   tokv[8];
        float wvv[8];
        #pragma unroll
        for (int v = 0; v < 8; ++v) {
            int ml = tile * 64 + m * 16 + v + 8 * hi;
            int sl = base + (ml < len ? ml : 0);
            tokv[v] = tok_id[sl];
            wvv[v]  = tok_w[sl];
        }
        #pragma unroll
        for (int n = 0; n < 4; ++n) {
            int col = nb + n * 16 + r;
            #pragma unroll
            for (int v = 0; v < 8; ++v) {
                int ml = tile * 64 + m * 16 + v + 8 * hi;
                if (ml < len)
                    atomicAdd(out + (size_t)tokv[v] * DIM + col,
                              wvv[v] * (acc[m][n][v] + bv[n]));
            }
        }
    }
}

// ---------------- host side ----------------
extern "C" void kernel_launch(void* const* d_in, const int* in_sizes, int n_in,
                              void* d_out, int out_size, void* d_ws, size_t ws_size,
                              hipStream_t stream) {
    const float* x      = (const float*)d_in[0];
    const float* gate_W = (const float*)d_in[1];
    const float* gate_b = (const float*)d_in[2];
    const float* ebias  = (const float*)d_in[3];
    const float* W1     = (const float*)d_in[4];
    const float* b1     = (const float*)d_in[5];
    const float* W2     = (const float*)d_in[6];
    const float* b2     = (const float*)d_in[7];
    const float* sW1    = (const float*)d_in[8];
    const float* sb1    = (const float*)d_in[9];
    const float* sW2    = (const float*)d_in[10];
    const float* sb2    = (const float*)d_in[11];
    float* out = (float*)d_out;

    char* ws = (char*)d_ws;
    __bf16* xbf   = (__bf16*)(ws + OFF_XBF);
    __bf16* w1t   = (__bf16*)(ws + OFF_W1T);
    __bf16* w2t   = (__bf16*)(ws + OFF_W2T);
    __bf16* hbuf  = (__bf16*)(ws + OFF_H);
    int*    tidx  = (int*)  (ws + OFF_TIDX);
    float*  tw    = (float*)(ws + OFF_TW);
    int*    tokid = (int*)  (ws + OFF_TOKID);
    float*  tokw  = (float*)(ws + OFF_TOKW);
    int*    counts    = (int*)  (ws + OFF_META);
    float*  P_sum     = (float*)(ws + OFF_META + 128);
    int*    seg_start = (int*)  (ws + OFF_META + 256);
    int*    seg_len   = (int*)  (ws + OFF_META + 384);
    int*    cursor    = (int*)  (ws + OFF_META + 512);

    hipMemsetAsync(d_out, 0, sizeof(float) * (size_t)out_size, stream);
    hipMemsetAsync(ws + OFF_META, 0, 640, stream);

    GeoMoE_cvt<<<(T_TOK * DIM) / 1024, 256, 0, stream>>>(x, xbf, T_TOK * DIM);

    dim3 tb(32, 8);
    GeoMoE_transpose<<<dim3(HID / 32, DIM / 32, NEXP), tb, 0, stream>>>(W1, w1t, DIM, HID);
    GeoMoE_transpose<<<dim3(HID / 32, DIM / 32, 1),    tb, 0, stream>>>(sW1, w1t + (size_t)NEXP * HID * DIM, DIM, HID);
    GeoMoE_transpose<<<dim3(DIM / 32, HID / 32, NEXP), tb, 0, stream>>>(W2, w2t, HID, DIM);
    GeoMoE_transpose<<<dim3(DIM / 32, HID / 32, 1),    tb, 0, stream>>>(sW2, w2t + (size_t)NEXP * DIM * HID, HID, DIM);

    GeoMoE_gate<<<T_TOK, 32, 0, stream>>>(x, gate_W, gate_b, ebias,
                                          tidx, tw, counts, P_sum, tokid, tokw);
    GeoMoE_scan<<<1, 32, 0, stream>>>(counts, P_sum, seg_start, seg_len, cursor, out);
    GeoMoE_scatter<<<(T_TOK * KTOP + 255) / 256, 256, 0, stream>>>(tidx, tw, seg_start,
                                                                   cursor, tokid, tokw);

    GeoMoE_ffn1<<<dim3(JOBS, HID / 64), 32, 0, stream>>>(xbf, w1t, b1, sb1,
                                                         seg_start, seg_len, tokid, hbuf);
    GeoMoE_ffn2<<<dim3(JOBS, DIM / 64), 32, 0, stream>>>(hbuf, w2t, b2, sb2,
                                                         seg_start, seg_len, tokid, tokw, out);
}
